// QLinear_83064667504675
// MI455X (gfx1250) — matrix operand order repaired and checked by reading, not compiled, not run on hardware
//
#include <hip/hip_runtime.h>

// ---- CDNA5 (gfx1250) fused int4-dequant GEMM + bias, f16 WMMA, f32 accum ----
// y[8192,11008] = x[8192,4096] @ ((W-8)*scale+zero)^T + bias
// Double-buffered LDS pipeline: global loads for tile k+1 overlap WMMA on tile k,
// one workgroup barrier per K-step.

typedef _Float16 v16h __attribute__((ext_vector_type(16)));
typedef _Float16 v8h  __attribute__((ext_vector_type(8)));
typedef float    v8f  __attribute__((ext_vector_type(8)));
typedef float    v4f  __attribute__((ext_vector_type(4)));
typedef int      v4i  __attribute__((ext_vector_type(4)));

#define TOKENS 8192
#define IN_F   4096
#define OUT_F  11008
#define GS     128

#define BM 128
#define BN 128
#define BK 64
#define LDA (BK + 8)   // f16 units, padded to dodge LDS bank conflicts
#define LDB (BK + 8)

static __device__ __forceinline__ v16h cat16(v8h lo, v8h hi) {
  v16h r;
#pragma unroll
  for (int i = 0; i < 8; ++i) { r[i] = lo[i]; r[i + 8] = hi[i]; }
  return r;
}

__global__ __launch_bounds__(256) void q4gemm_wmma(
    const float* __restrict__ x,      // [TOKENS, IN_F] f32
    const int*   __restrict__ w,      // [OUT_F, IN_F] int4 codes in int32
    const float* __restrict__ sz,     // [IN_F/GS, OUT_F, 2] (scale, zero)
    const float* __restrict__ bias,   // [OUT_F]
    float*       __restrict__ y)      // [TOKENS, OUT_F]
{
  __shared__ _Float16 As[2][BM * LDA];   // x tile, f16 (ping-pong)
  __shared__ _Float16 Bs[2][BN * LDB];   // dequantized weight tile, f16 (ping-pong)

  const int tid   = threadIdx.x;
  const int lane  = tid & 31;            // wave32
  const int wave  = tid >> 5;            // 8 waves / block
  const int waveM = wave & 1;            // 2 waves along M
  const int waveN = wave >> 1;           // 4 waves along N

  const int bn0 = blockIdx.x * BN;
  const int bm0 = blockIdx.y * BM;

  v8f acc[4][2] = {};                    // wave computes 64(M) x 32(N)

  // Cooperative tile loads: thread -> (row, 32-wide k chunk)
  const int lrow = tid >> 1;             // 0..127
  const int lk   = (tid & 1) << 5;       // 0 or 32

  const float* xrow = x + (size_t)(bm0 + lrow) * IN_F + lk;
  const int*   wrow = w + (size_t)(bn0 + lrow) * IN_F + lk;

  // Register staging for the in-flight global tile
  v4f   xs[8];
  v4i   qs[8];
  float sc, zp;

  auto LOAD = [&](int k0) {
#pragma unroll
    for (int v = 0; v < 8; ++v) {
      xs[v] = *(const v4f*)(xrow + k0 + v * 4);   // global_load_b128
      qs[v] = *(const v4i*)(wrow + k0 + v * 4);   // global_load_b128
    }
    const int g = k0 / GS;                         // BK=64 < GS=128: one group/tile
    const size_t szi = ((size_t)g * OUT_F + (size_t)(bn0 + lrow)) * 2;
    sc = sz[szi + 0];
    zp = sz[szi + 1];
  };

  auto STORE = [&](int buf) {
    _Float16* ap = &As[buf][lrow * LDA + lk];
    _Float16* bp = &Bs[buf][lrow * LDB + lk];
#pragma unroll
    for (int v = 0; v < 8; ++v) {
#pragma unroll
      for (int e = 0; e < 4; ++e) {
        ap[v * 4 + e] = (_Float16)xs[v][e];
        bp[v * 4 + e] = (_Float16)((float)(qs[v][e] - 8) * sc + zp);
      }
    }
  };

  auto COMPUTE = [&](int buf) {
#pragma unroll
    for (int kk = 0; kk < BK; kk += 32) {
      // A frag (16-bit A 16x32 layout): m = lane%16, K in {b..b+7} u {b+16..b+23},
      // b = 8*(lane/16)
      const int abase = kk + ((lane >> 4) << 3);
      // B frag (column per lane): n = lane%16, K in {b..b+7} u {b+16..b+23}, b = 8*(lane/16)
      const int bbase = kk + ((lane >> 4) << 3);
      const int mr = (waveM << 6) + (lane & 15);
      const int nr = (waveN << 5) + (lane & 15);

      v16h a[4];
#pragma unroll
      for (int mi = 0; mi < 4; ++mi) {
        const _Float16* p = &As[buf][(mr + mi * 16) * LDA + abase];
        a[mi] = cat16(*(const v8h*)p, *(const v8h*)(p + 16));
      }
      v16h b[2];
#pragma unroll
      for (int ni = 0; ni < 2; ++ni) {
        const _Float16* p = &Bs[buf][(nr + ni * 16) * LDB + bbase];
        b[ni] = cat16(*(const v8h*)p, *(const v8h*)(p + 16));
      }

#pragma unroll
      for (int mi = 0; mi < 4; ++mi)
#pragma unroll
        for (int ni = 0; ni < 2; ++ni)
          acc[mi][ni] = __builtin_amdgcn_wmma_f32_16x16x32_f16(
              /*neg_a=*/false, a[mi], /*neg_b=*/false, b[ni],
              /*c_mod=*/(short)0, acc[mi][ni],
              /*reuse_a=*/false, /*reuse_b=*/false);
    }
  };

  // ---- Pipeline prologue ----
  LOAD(0);
  STORE(0);
  __syncthreads();

  int cur = 0;
  for (int k0 = 0; k0 < IN_F; k0 += BK) {
    const bool hasNext = (k0 + BK) < IN_F;

    if (hasNext) {
      LOAD(k0 + BK);                       // overlaps with WMMA below
      if (k0 + 2 * BK < IN_F) {            // warm L2 two tiles ahead
        __builtin_prefetch(xrow + k0 + 2 * BK, 0, 0);  // global_prefetch_b8
        __builtin_prefetch(wrow + k0 + 2 * BK, 0, 0);
      }
    }

    COMPUTE(cur);                          // reads buf cur

    if (hasNext) {
      STORE(cur ^ 1);                      // writes other buffer: no conflict with
                                           // COMPUTE(cur); prior readers of cur^1
                                           // were fenced by last iteration's barrier
      __syncthreads();                     // single barrier per K-step
      cur ^= 1;
    }
  }

  // ---- Epilogue: C/D layout — lane 0-15: M=r, lane 16-31: M=r+8; N=lane%16 ----
  const int m0 = bm0 + (waveM << 6) + ((lane >> 4) << 3);
  const int n0 = bn0 + (waveN << 5) + (lane & 15);
#pragma unroll
  for (int ni = 0; ni < 2; ++ni) {
    const int ncol = n0 + ni * 16;
    const float bv = bias[ncol];
#pragma unroll
    for (int mi = 0; mi < 4; ++mi) {
#pragma unroll
      for (int r = 0; r < 8; ++r) {
        const int mrow = m0 + mi * 16 + r;
        y[(size_t)mrow * OUT_F + ncol] = acc[mi][ni][r] + bv;
      }
    }
  }
}

extern "C" void kernel_launch(void* const* d_in, const int* in_sizes, int n_in,
                              void* d_out, int out_size, void* d_ws, size_t ws_size,
                              hipStream_t stream) {
  const float* x    = (const float*)d_in[0];
  const int*   w    = (const int*)d_in[1];
  const float* sz   = (const float*)d_in[2];
  const float* bias = (const float*)d_in[3];
  float*       y    = (float*)d_out;

  dim3 grid(OUT_F / BN, TOKENS / BM);   // (86, 64)
  q4gemm_wmma<<<grid, 256, 0, stream>>>(x, w, sz, bias, y);
}
